// TemporalMHA_712964571497
// MI455X (gfx1250) — matrix operand order repaired and checked by reading, not compiled, not run on hardware
//
#include <hip/hip_runtime.h>
#include <hip/hip_bf16.h>
#include <stdint.h>
#include <math.h>

#define B_  2
#define L_  2048
#define D_  1024
#define H_  16
#define DK_ 64
#define M_  (B_*L_)   // 4096

typedef unsigned short u16;
typedef __attribute__((ext_vector_type(16))) __bf16 bf16x16;
typedef __attribute__((ext_vector_type(8)))  float  f32x8;
typedef __attribute__((ext_vector_type(4)))  unsigned int u32x4;
typedef __attribute__((ext_vector_type(8)))  int i32x8;
typedef __attribute__((ext_vector_type(4)))  int i32x4;

union BFrag { uint4 u[2]; bf16x16 v; };

// Toolchain discrimination: therock headers => clang-23 6-arg TDM builtin.
#if __has_include(<hip/amd_detail/amd_gfx1250_TDM.h>)
#define TDM_SIX_ARG 1
#endif
#if __has_builtin(__builtin_amdgcn_tensor_load_to_lds)
#define USE_TDM 1
#endif
#if __has_builtin(__builtin_amdgcn_s_wait_tensorcnt)
#define WAIT_TENSOR(n) __builtin_amdgcn_s_wait_tensorcnt(n)
#else
#define WAIT_TENSOR(n) asm volatile("s_wait_tensorcnt %0" :: "i"(n) : "memory")
#endif

__device__ __forceinline__ u16 f2bf(float f) {
  unsigned int u = __float_as_uint(f);
  u += 0x7FFFu + ((u >> 16) & 1u);   // round-to-nearest-even
  return (u16)(u >> 16);
}

#ifdef USE_TDM
// TDM: DMA a 128(row) x 32(halfs) bf16 tile from a [rows, K] row-major tensor
// into LDS with 64B data + 16B pad per row (row stride 40 halfs = 80B).
__device__ __forceinline__ void tdm_load_tile_128x32(
    const u16* gsrc, unsigned int ldsaddr, unsigned int K)
{
  unsigned long long ga = (unsigned long long)(uintptr_t)gsrc;
  u32x4 g0;
  g0[0] = 1u;                                   // count=1, user-mode, no gather
  g0[1] = ldsaddr;                              // lds_addr
  g0[2] = (unsigned int)ga;                     // global_addr[31:0]
  g0[3] = (unsigned int)((ga >> 32) & 0x1FFFFFFull) | (2u << 30); // ga[56:32] | type=2
  i32x8 g1;
  // data_size=1(2B) | pad_enable | pad_interval=3(16 dwords=64B) | pad_amount=3(4 dwords=16B)
  g1[0] = (int)((1u << 16) | (1u << 20) | (3u << 22) | (3u << 25));
  g1[1] = (int)((K & 0xFFFFu) << 16);                           // abar=0 | tensor_dim0 lo
  g1[2] = (int)(((K >> 16) & 0xFFFFu) | (((unsigned)M_ & 0xFFFFu) << 16)); // dim0 hi | dim1 lo
  g1[3] = (int)((((unsigned)M_ >> 16) & 0xFFFFu) | (32u << 16)); // dim1 hi | tile_dim0=32
  g1[4] = (int)128u;                                            // tile_dim1=128 | tile_dim2=0
  g1[5] = (int)K;                                               // tensor_dim0_stride lo
  g1[6] = 0;                                                    // stride hi | dim1_stride lo
  g1[7] = 0;
  i32x4 gz = {0, 0, 0, 0};
#ifdef TDM_SIX_ARG
  i32x8 gz8 = {0, 0, 0, 0, 0, 0, 0, 0};
  __builtin_amdgcn_tensor_load_to_lds(g0, g1, gz, gz, gz8, 0);
#else
  __builtin_amdgcn_tensor_load_to_lds(g0, g1, gz, gz, 0);
#endif
}
#endif

// ---------------------------------------------------------------------------
// f32 -> bf16 conversion
// ---------------------------------------------------------------------------
__global__ void cvt_f32_bf16(const float* __restrict__ s, u16* __restrict__ d, int n) {
  int i = blockIdx.x * blockDim.x + threadIdx.x;
  int stride = gridDim.x * blockDim.x;
  for (; i < n; i += stride) d[i] = f2bf(s[i]);
}

// denom = max(max_b(t[b,L-1]-t[b,0]), 1.0)  (timestamps sorted ascending)
__global__ void calc_denom(const float* __restrict__ ts, float* __restrict__ out) {
  float m = 1.0f;
  for (int b = 0; b < B_; ++b)
    m = fmaxf(m, ts[(size_t)b * L_ + (L_ - 1)] - ts[(size_t)b * L_]);
  out[0] = m;
}

// ---------------------------------------------------------------------------
// GEMM: out[m,n] = sum_k A[m,k]*W[n,k] + bias[n]
//   A: [M,K] bf16 row-major;  W: [N,K] bf16 row-major (== B-matrix in [N][K])
//   mode 0: store bf16 head-split [B,H,L,DK];  mode 1: store f32 flat [M,N]
// 256 threads (8 waves), tile 128x128x32, wave tile 32x64 (2x4 WMMA).
// Data movement: TDM tensor_load_to_lds, double-buffered LDS.
// ---------------------------------------------------------------------------
__global__ __launch_bounds__(256) void gemm_bf16_wmma(
    const u16* __restrict__ A, const u16* __restrict__ W,
    const float* __restrict__ bias, void* __restrict__ outp,
    int M, int N, int K, int mode)
{
  __shared__ u16 As[2][128 * 40];
  __shared__ u16 Bs[2][128 * 40];

  const int tid  = threadIdx.x;
  const int lane = tid & 31;
  const int wave = tid >> 5;
  const int wm = wave >> 1, wn = wave & 1;
  const int m0 = blockIdx.y * 128;
  const int n0 = blockIdx.x * 128;
  const int lm = lane & 15, lg = lane >> 4;

  f32x8 acc[2][4] = {};

#ifdef USE_TDM
  const int nk = K >> 5;
  const u16* Abase = A + (size_t)m0 * K;
  const u16* Wbase = W + (size_t)n0 * K;
  if (wave == 0) {
    tdm_load_tile_128x32(Abase, (unsigned int)(uintptr_t)&As[0][0], (unsigned)K);
    tdm_load_tile_128x32(Wbase, (unsigned int)(uintptr_t)&Bs[0][0], (unsigned)K);
  }
  for (int j = 0; j < nk; ++j) {
    const int cur = j & 1;
    const int nxt = cur ^ 1;
    if (wave == 0) {
      if (j + 1 < nk) {
        // prefetch next tile via TDM while current tile is consumed
        tdm_load_tile_128x32(Abase + (size_t)(j + 1) * 32, (unsigned int)(uintptr_t)&As[nxt][0], (unsigned)K);
        tdm_load_tile_128x32(Wbase + (size_t)(j + 1) * 32, (unsigned int)(uintptr_t)&Bs[nxt][0], (unsigned)K);
        WAIT_TENSOR(2);   // TDM in-order per wave: <=2 outstanding => tile j landed
      } else {
        WAIT_TENSOR(0);
      }
    }
    __syncthreads();      // release all waves to read buffer `cur`

    BFrag af[2], bfr[4];
#pragma unroll
    for (int tm = 0; tm < 2; ++tm) {            // A-frag: row=lane%16, K chunks by lane/16
      int r = wm * 32 + tm * 16 + lm;
      af[tm].u[0] = *(const uint4*)&As[cur][r * 40 + lg * 8];
      af[tm].u[1] = *(const uint4*)&As[cur][r * 40 + 16 + lg * 8];
    }
#pragma unroll
    for (int tn = 0; tn < 4; ++tn) {            // B-frag: col=lane%16, K chunks by lane/16
      int c = wn * 64 + tn * 16 + lm;
      bfr[tn].u[0] = *(const uint4*)&Bs[cur][c * 40 + lg * 8];
      bfr[tn].u[1] = *(const uint4*)&Bs[cur][c * 40 + 16 + lg * 8];
    }
#pragma unroll
    for (int tm = 0; tm < 2; ++tm)
#pragma unroll
      for (int tn = 0; tn < 4; ++tn)
        acc[tm][tn] = __builtin_amdgcn_wmma_f32_16x16x32_bf16(
            false, af[tm].v, false, bfr[tn].v, (short)0, acc[tm][tn], false, false);
    __syncthreads();      // buffer `cur` drained; safe to DMA into it next iter
  }
#else
  // Fallback staging path (manual global->LDS), single buffer.
  const int lrow = tid >> 1;
  const int lseg = (tid & 1) * 16;
  for (int k0 = 0; k0 < K; k0 += 32) {
    {
      const uint4* ga = (const uint4*)(A + (size_t)(m0 + lrow) * K + k0 + lseg);
      uint4 a0 = ga[0], a1 = ga[1];
      *(uint4*)&As[0][lrow * 40 + lseg]     = a0;
      *(uint4*)&As[0][lrow * 40 + lseg + 8] = a1;
      const uint4* gb = (const uint4*)(W + (size_t)(n0 + lrow) * K + k0 + lseg);
      uint4 b0 = gb[0], b1 = gb[1];
      *(uint4*)&Bs[0][lrow * 40 + lseg]     = b0;
      *(uint4*)&Bs[0][lrow * 40 + lseg + 8] = b1;
      if (k0 + 32 < K) {
        __builtin_prefetch(A + (size_t)(m0 + lrow) * K + k0 + 32 + lseg, 0, 1);
        __builtin_prefetch(W + (size_t)(n0 + lrow) * K + k0 + 32 + lseg, 0, 1);
      }
    }
    __syncthreads();
    BFrag af[2], bfr[4];
#pragma unroll
    for (int tm = 0; tm < 2; ++tm) {
      int r = wm * 32 + tm * 16 + lm;
      af[tm].u[0] = *(const uint4*)&As[0][r * 40 + lg * 8];
      af[tm].u[1] = *(const uint4*)&As[0][r * 40 + 16 + lg * 8];
    }
#pragma unroll
    for (int tn = 0; tn < 4; ++tn) {
      int c = wn * 64 + tn * 16 + lm;
      bfr[tn].u[0] = *(const uint4*)&Bs[0][c * 40 + lg * 8];
      bfr[tn].u[1] = *(const uint4*)&Bs[0][c * 40 + 16 + lg * 8];
    }
#pragma unroll
    for (int tm = 0; tm < 2; ++tm)
#pragma unroll
      for (int tn = 0; tn < 4; ++tn)
        acc[tm][tn] = __builtin_amdgcn_wmma_f32_16x16x32_bf16(
            false, af[tm].v, false, bfr[tn].v, (short)0, acc[tm][tn], false, false);
    __syncthreads();
  }
#endif

  // epilogue: add bias, scatter
#pragma unroll
  for (int tn = 0; tn < 4; ++tn) {
    int n = n0 + wn * 64 + tn * 16 + lm;
    float bn = bias[n];
#pragma unroll
    for (int tm = 0; tm < 2; ++tm) {
#pragma unroll
      for (int v = 0; v < 8; ++v) {             // C-layout: M = v + 8*(lane/16)
        int m = m0 + wm * 32 + tm * 16 + v + (lg << 3);
        float val = acc[tm][tn][v] + bn;
        if (mode == 0) {
          int b = m >> 11, l = m & (L_ - 1);
          int h = n >> 6,  dk = n & (DK_ - 1);
          ((u16*)outp)[(((size_t)b * H_ + h) * L_ + l) * DK_ + dk] = f2bf(val);
        } else {
          ((float*)outp)[(size_t)m * N + n] = val;
        }
      }
    }
  }
}

// ---------------------------------------------------------------------------
// Flash-style causal attention with temporal log-decay bias.
// Block = 128 threads (4 waves); wave owns 16 q rows; kv in blocks of 32.
// ---------------------------------------------------------------------------
__global__ __launch_bounds__(128) void attn_wmma(
    const u16* __restrict__ Q, const u16* __restrict__ Km, const u16* __restrict__ Vm,
    const float* __restrict__ ts, const unsigned char* __restrict__ pad,
    const float* __restrict__ lamp, const float* __restrict__ denomp,
    u16* __restrict__ ctx)
{
  __shared__ u16 Plds[4][16 * 40];   // per-wave P re-layout scratch
  __shared__ u16 Vt[4][64 * 40];     // per-wave V^T tile [dk][kv]

  const int lane = threadIdx.x & 31;
  const int wave = threadIdx.x >> 5;
  const int h = blockIdx.y, b = blockIdx.z;
  const int qbase = blockIdx.x * 64 + wave * 16;
  const size_t bh = (size_t)b * H_ + h;
  const u16* Qp = Q  + bh * L_ * DK_;
  const u16* Kp = Km + bh * L_ * DK_;
  const u16* Vp = Vm + bh * L_ * DK_;

  const int lm = lane & 15, lg = lane >> 4;
  const float lam   = fabsf(lamp[0]);
  const float denom = denomp[0];
  const float sc = 0.125f;  // 1/sqrt(DK)

  BFrag qf[2];
  {
    const u16* qrow = Qp + (size_t)(qbase + lm) * DK_;
    qf[0].u[0] = *(const uint4*)(qrow + lg * 8);
    qf[0].u[1] = *(const uint4*)(qrow + 16 + lg * 8);
    qf[1].u[0] = *(const uint4*)(qrow + 32 + lg * 8);
    qf[1].u[1] = *(const uint4*)(qrow + 48 + lg * 8);
  }

  float tq[8];
#pragma unroll
  for (int v = 0; v < 8; ++v) tq[v] = ts[(size_t)b * L_ + qbase + v + (lg << 3)];

  float rowmax[8], rowsum[8];
#pragma unroll
  for (int v = 0; v < 8; ++v) { rowmax[v] = -INFINITY; rowsum[v] = 0.f; }
  f32x8 oacc[4] = {};

  const int nkv = (qbase + 15) / 32 + 1;   // causal: only kv blocks with kb <= q_max
  for (int jb = 0; jb < nkv; ++jb) {
    const int kb = jb * 32;

    // K fragments: B-layout of K^T (contraction over DK), 2 kv subtiles x 2 dk halves
    BFrag kf[2][2];
#pragma unroll
    for (int jn = 0; jn < 2; ++jn) {
      const u16* krow = Kp + (size_t)(kb + jn * 16 + lm) * DK_;
#pragma unroll
      for (int hh = 0; hh < 2; ++hh) {
        kf[jn][hh].u[0] = *(const uint4*)(krow + hh * 32 + lg * 8);
        kf[jn][hh].u[1] = *(const uint4*)(krow + hh * 32 + 16 + lg * 8);
      }
    }

    f32x8 s[2] = {};
#pragma unroll
    for (int jn = 0; jn < 2; ++jn) {
      s[jn] = __builtin_amdgcn_wmma_f32_16x16x32_bf16(false, qf[0].v, false, kf[jn][0].v, (short)0, s[jn], false, false);
      s[jn] = __builtin_amdgcn_wmma_f32_16x16x32_bf16(false, qf[1].v, false, kf[jn][1].v, (short)0, s[jn], false, false);
    }

    // Stage V tile transposed into LDS: Vt[dk][kv] (32 kv x 64 dk)
    {
      const int kvq  = (lane & 7) * 4;
      const int dseg = (lane >> 3) * 16;
      union { uint4 q[2]; u16 hlf[16]; } rowj[4];
#pragma unroll
      for (int j = 0; j < 4; ++j) {
        const uint4* vp = (const uint4*)(Vp + (size_t)(kb + kvq + j) * DK_ + dseg);
        rowj[j].q[0] = vp[0];
        rowj[j].q[1] = vp[1];
      }
#pragma unroll
      for (int c = 0; c < 16; ++c) {
        unsigned long long w =
              (unsigned long long)rowj[0].hlf[c]
            | ((unsigned long long)rowj[1].hlf[c] << 16)
            | ((unsigned long long)rowj[2].hlf[c] << 32)
            | ((unsigned long long)rowj[3].hlf[c] << 48);
        *(unsigned long long*)&Vt[wave][(dseg + c) * 40 + kvq] = w;
      }
    }

    // Bias, masks, online softmax (f32). Row M = v + 8*(lane/16), col = kb + jn*16 + lm.
    float tk0 = ts[(size_t)b * L_ + kb + lm];
    float tk1 = ts[(size_t)b * L_ + kb + 16 + lm];
    int pv0 = pad[(size_t)b * L_ + kb + lm] ? 1 : 0;
    int pv1 = pad[(size_t)b * L_ + kb + 16 + lm] ? 1 : 0;

    float p0[8], p1[8], alpha[8];
#pragma unroll
    for (int v = 0; v < 8; ++v) {
      int q = qbase + v + (lg << 3);
      int kv0 = kb + lm, kv1 = kb + 16 + lm;
      float s0 = s[0][v] * sc - lam * log1pf(fabsf(tq[v] - tk0) / denom);
      float s1 = s[1][v] * sc - lam * log1pf(fabsf(tq[v] - tk1) / denom);
      s0 = (kv0 <= q && pv0) ? s0 : -INFINITY;
      s1 = (kv1 <= q && pv1) ? s1 : -INFINITY;
      float t = fmaxf(s0, s1);
#pragma unroll
      for (int msk = 1; msk < 16; msk <<= 1) t = fmaxf(t, __shfl_xor(t, msk, 32));
      float nm = fmaxf(rowmax[v], t);
      alpha[v] = __expf(rowmax[v] - nm);
      rowmax[v] = nm;
      float e0 = __expf(s0 - nm);
      float e1 = __expf(s1 - nm);
      p0[v] = e0; p1[v] = e1;
      float ps = e0 + e1;
#pragma unroll
      for (int msk = 1; msk < 16; msk <<= 1) ps += __shfl_xor(ps, msk, 32);
      rowsum[v] = rowsum[v] * alpha[v] + ps;
    }
#pragma unroll
    for (int tn = 0; tn < 4; ++tn)
#pragma unroll
      for (int v = 0; v < 8; ++v) oacc[tn][v] *= alpha[v];

    // P: C-layout -> A-layout via per-wave LDS
#pragma unroll
    for (int v = 0; v < 8; ++v) {
      int r = v + (lg << 3);
      Plds[wave][r * 40 + lm]      = f2bf(p0[v]);
      Plds[wave][r * 40 + 16 + lm] = f2bf(p1[v]);
    }
    asm volatile("s_wait_dscnt 0x0" ::: "memory");  // same-wave DS ordering

    BFrag pf;
    pf.u[0] = *(const uint4*)&Plds[wave][lm * 40 + lg * 8];
    pf.u[1] = *(const uint4*)&Plds[wave][lm * 40 + 16 + lg * 8];

#pragma unroll
    for (int tn = 0; tn < 4; ++tn) {
      BFrag vf;
      int n = tn * 16 + lm;
      vf.u[0] = *(const uint4*)&Vt[wave][n * 40 + lg * 8];
      vf.u[1] = *(const uint4*)&Vt[wave][n * 40 + 16 + lg * 8];
      oacc[tn] = __builtin_amdgcn_wmma_f32_16x16x32_bf16(
          false, pf.v, false, vf.v, (short)0, oacc[tn], false, false);
    }
  }

  // normalize + store context bf16 [B,L,D]
#pragma unroll
  for (int v = 0; v < 8; ++v) {
    int q = qbase + v + (lg << 3);
    float r = rowsum[v];
    float inv = (r > 0.f) ? (1.f / r) : 0.f;   // nan_to_num guard
#pragma unroll
    for (int tn = 0; tn < 4; ++tn) {
      int dk = tn * 16 + lm;
      ctx[((size_t)b * L_ + q) * D_ + h * DK_ + dk] = f2bf(oacc[tn][v] * inv);
    }
  }
}

// ---------------------------------------------------------------------------
extern "C" void kernel_launch(void* const* d_in, const int* in_sizes, int n_in,
                              void* d_out, int out_size, void* d_ws, size_t ws_size,
                              hipStream_t stream) {
  (void)in_sizes; (void)n_in; (void)out_size; (void)ws_size;
  const float* x   = (const float*)d_in[0];
  const float* ts  = (const float*)d_in[1];
  /* d_in[2] decay_values: unused by the reference */
  const unsigned char* pad = (const unsigned char*)d_in[3];
  const float* Wq = (const float*)d_in[4];
  const float* bq = (const float*)d_in[5];
  const float* Wk = (const float*)d_in[6];
  const float* bk = (const float*)d_in[7];
  const float* Wv = (const float*)d_in[8];
  const float* bv = (const float*)d_in[9];
  const float* Wo = (const float*)d_in[10];
  const float* bo = (const float*)d_in[11];
  const float* lam = (const float*)d_in[12];

  char* ws = (char*)d_ws;
  u16* xb  = (u16*)(ws);                       //  8 MB: x bf16 [4096,1024]
  u16* wqb = (u16*)(ws + (size_t)8  * 1024 * 1024);
  u16* wkb = (u16*)(ws + (size_t)10 * 1024 * 1024);
  u16* wvb = (u16*)(ws + (size_t)12 * 1024 * 1024);
  u16* wob = (u16*)(ws + (size_t)14 * 1024 * 1024);
  u16* Qb  = (u16*)(ws + (size_t)16 * 1024 * 1024);  // [B,H,L,DK] bf16
  u16* Kb  = (u16*)(ws + (size_t)24 * 1024 * 1024);
  u16* Vb  = (u16*)(ws + (size_t)32 * 1024 * 1024);
  u16* ctx = (u16*)(ws + (size_t)40 * 1024 * 1024);  // [B,L,D] bf16
  float* dn = (float*)(ws + (size_t)48 * 1024 * 1024);

  cvt_f32_bf16<<<1024, 256, 0, stream>>>(x,  xb,  M_ * D_);
  cvt_f32_bf16<<<512,  256, 0, stream>>>(Wq, wqb, D_ * D_);
  cvt_f32_bf16<<<512,  256, 0, stream>>>(Wk, wkb, D_ * D_);
  cvt_f32_bf16<<<512,  256, 0, stream>>>(Wv, wvb, D_ * D_);
  cvt_f32_bf16<<<512,  256, 0, stream>>>(Wo, wob, D_ * D_);
  calc_denom<<<1, 1, 0, stream>>>(ts, dn);

  dim3 gg(D_ / 128, M_ / 128);
  gemm_bf16_wmma<<<gg, 256, 0, stream>>>(xb, wqb, bq, Qb, M_, D_, D_, 0);
  gemm_bf16_wmma<<<gg, 256, 0, stream>>>(xb, wkb, bk, Kb, M_, D_, D_, 0);
  gemm_bf16_wmma<<<gg, 256, 0, stream>>>(xb, wvb, bv, Vb, M_, D_, D_, 0);

  dim3 ga(L_ / 64, H_, B_);
  attn_wmma<<<ga, 128, 0, stream>>>(Qb, Kb, Vb, ts, pad, lam, dn, ctx);

  gemm_bf16_wmma<<<gg, 256, 0, stream>>>(ctx, wob, bo, d_out, M_, D_, D_, 1);
}
